// MultiHeadAttention_50929722196632
// MI455X (gfx1250) — compile-verified, matrix operand order checked by reading, not run on hardware
//
#include <hip/hip_runtime.h>

// ---------------------------------------------------------------------------
// MI455X (gfx1250) attention: bf16 WMMA (v_wmma_f32_16x16x32_bf16), fp32 accum.
// ~137 GFLOP total, compute-bound => matrix cores. wave32, 16x16x32 tiles.
// Workspace layout (needs 64 MB): Q(16MB) K(16MB) Vt(16MB) Ctx(16MB), all bf16.
// ---------------------------------------------------------------------------

typedef __attribute__((ext_vector_type(16))) __bf16 v16bf;
typedef __attribute__((ext_vector_type(8)))  __bf16 v8bf;
typedef __attribute__((ext_vector_type(4)))  __bf16 v4bf;
typedef __attribute__((ext_vector_type(8)))  float  v8f;
typedef __attribute__((ext_vector_type(4)))  float  v4f;

union Frag { v16bf v; v8bf h[2]; };

__device__ __forceinline__ v8f zero8() {
  v8f z;
#pragma unroll
  for (int e = 0; e < 8; ++e) z[e] = 0.0f;
  return z;
}

// ---------------------------------------------------------------------------
// Generic tiled GEMM: C[M,N] = A[M,K] * B[K,N].
//  - B always fp32 weights, converted to bf16 while staged transposed in LDS.
//  - A either fp32 (converted in staging) or bf16.
//  - Output fp32, bf16, or bf16-transposed-per-batch (for Vt).
// Block: 256 threads = 8 waves, block tile 128x128, wave tile 64x32 (4x2 WMMA).
// Double-buffered LDS: global loads for tile i+1 issue before the WMMAs of
// tile i, LDS stores after, single barrier per K-step.
// ---------------------------------------------------------------------------
#define LDA_STRIDE 40   // 32 + 8 bf16 pad (16B-aligned rows, conflict-free)

template <bool A_BF16, bool OUT_BF16, bool OUT_T>
__global__ __launch_bounds__(256)
void wmma_gemm(const void* __restrict__ Ain, const float* __restrict__ Bw,
               void* __restrict__ Cout, int M, int N, int K, int S_T)
{
  __shared__ __align__(16) __bf16 lA[2][128 * LDA_STRIDE];
  __shared__ __align__(16) __bf16 lB[2][128 * LDA_STRIDE];   // stored [n][k]

  const int tid  = threadIdx.x;
  const int lane = tid & 31;
  const int w    = tid >> 5;
  const int wm   = w >> 2;                 // 0..1 -> 64 rows each
  const int wn   = w & 3;                  // 0..3 -> 32 cols each
  const int blockRow = blockIdx.y * 128;
  const int blockCol = blockIdx.x * 128;
  const int lr = lane & 15;
  const int lh = (lane >> 4) << 3;         // 0 or 8

  // Per-thread staging registers (4 x 4-wide units each for A and B)
  v4bf aH[4];
  v4f  aF[4];
  v4f  bF[4];

  auto load_stage = [&](int k0) {
#pragma unroll
    for (int s = 0; s < 4; ++s) {
      const int u   = tid + s * 256;
      const int row = u >> 3;              // A: 128 rows x 32 cols
      const int col = (u & 7) << 2;
      if constexpr (A_BF16) {
        aH[s] = *(const v4bf*)((const __bf16*)Ain +
                               (size_t)(blockRow + row) * K + k0 + col);
      } else {
        aF[s] = *(const v4f*)((const float*)Ain +
                              (size_t)(blockRow + row) * K + k0 + col);
      }
      const int kr = u >> 5;               // B: 32 k-rows x 128 cols
      const int nc = (u & 31) << 2;
      bF[s] = *(const v4f*)(Bw + (size_t)(k0 + kr) * N + blockCol + nc);
    }
  };
  auto store_stage = [&](int buf) {
#pragma unroll
    for (int s = 0; s < 4; ++s) {
      const int u   = tid + s * 256;
      const int row = u >> 3;
      const int col = (u & 7) << 2;
      v4bf h;
      if constexpr (A_BF16) {
        h = aH[s];
      } else {
        h[0] = (__bf16)aF[s][0]; h[1] = (__bf16)aF[s][1];
        h[2] = (__bf16)aF[s][2]; h[3] = (__bf16)aF[s][3];
      }
      *(v4bf*)&lA[buf][row * LDA_STRIDE + col] = h;
      const int kr = u >> 5;
      const int nc = (u & 31) << 2;
      lB[buf][(nc + 0) * LDA_STRIDE + kr] = (__bf16)bF[s][0];
      lB[buf][(nc + 1) * LDA_STRIDE + kr] = (__bf16)bF[s][1];
      lB[buf][(nc + 2) * LDA_STRIDE + kr] = (__bf16)bF[s][2];
      lB[buf][(nc + 3) * LDA_STRIDE + kr] = (__bf16)bF[s][3];
    }
  };

  v8f acc[4][2];
#pragma unroll
  for (int i = 0; i < 4; ++i)
#pragma unroll
    for (int j = 0; j < 2; ++j) acc[i][j] = zero8();

  load_stage(0);
  store_stage(0);
  __syncthreads();

  for (int k0 = 0; k0 < K; k0 += 32) {
    const int  cur     = (k0 >> 5) & 1;
    const bool hasNext = (k0 + 32) < K;
    if (hasNext) load_stage(k0 + 32);      // overlap with WMMAs below

    Frag a[4], b[2];
#pragma unroll
    for (int i = 0; i < 4; ++i) {
      const __bf16* p = &lA[cur][(wm * 64 + i * 16 + lr) * LDA_STRIDE + lh];
      a[i].h[0] = *(const v8bf*)p;
      a[i].h[1] = *(const v8bf*)(p + 16);
    }
#pragma unroll
    for (int j = 0; j < 2; ++j) {
      const __bf16* p = &lB[cur][(wn * 32 + j * 16 + lr) * LDA_STRIDE + lh];
      b[j].h[0] = *(const v8bf*)p;
      b[j].h[1] = *(const v8bf*)(p + 16);
    }
#pragma unroll
    for (int i = 0; i < 4; ++i)
#pragma unroll
      for (int j = 0; j < 2; ++j)
        acc[i][j] = __builtin_amdgcn_wmma_f32_16x16x32_bf16(
            false, a[i].v, false, b[j].v, (short)0, acc[i][j], false, false);

    if (hasNext) store_stage(cur ^ 1);
    __syncthreads();
  }

  // ---- epilogue: C/D layout: VGPR r -> row r + 8*(lane>>4), col lane&15 ----
#pragma unroll
  for (int i = 0; i < 4; ++i)
#pragma unroll
    for (int j = 0; j < 2; ++j)
#pragma unroll
      for (int r = 0; r < 8; ++r) {
        const int row = blockRow + wm * 64 + i * 16 + r + lh;
        const int col = blockCol + wn * 32 + j * 16 + lr;
        const float val = acc[i][j][r];
        if constexpr (OUT_T) {   // Vt[b][col][s] for per-batch transposed V
          const int bb = row / S_T, sr = row % S_T;
          ((__bf16*)Cout)[((size_t)bb * N + col) * S_T + sr] = (__bf16)val;
        } else if constexpr (OUT_BF16) {
          ((__bf16*)Cout)[(size_t)row * N + col] = (__bf16)val;
        } else {
          ((float*)Cout)[(size_t)row * N + col] = val;
        }
      }
}

// ---------------------------------------------------------------------------
// Fused attention: per block = one batch b, 16 query rows.
//  Phase 1: scores[16][2048] = (Q tile @ K^T) * 1/sqrt(64), fp32 in LDS.
//           Wave w owns key tiles [w*16, w*16+16), 8 accumulators at a time:
//           each Q A-fragment is loaded from LDS once per k-step and feeds
//           8 independent WMMAs (B-frags stream straight from global K,
//           whose row-major [s][d] layout IS the B^T fragment layout).
//  Phase 2: exact softmax over the 2048-wide rows in LDS.
//  Phase 3: Ctx tile = P @ V with Vt[b][d][s]; 8 accumulators per wave,
//           P A-fragment loaded once per k-step.
// Dynamic LDS ~226 KB (< 320 KB/WGP).
// ---------------------------------------------------------------------------
#define LQ_STRIDE 1032   // 1024 + 8 bf16 pad
#define LS_STRIDE 2052   // 2048 + 4 f32 pad
#define LP_STRIDE 2056   // 2048 + 8 bf16 pad
#define ATT_SMEM_BYTES (16*LQ_STRIDE*2 + 16*LS_STRIDE*4 + 16*LP_STRIDE*2 + 16*16*4)

__global__ __launch_bounds__(256, 1)
void attn_fused(const __bf16* __restrict__ Q, const __bf16* __restrict__ Km,
                const __bf16* __restrict__ Vt, __bf16* __restrict__ Ctx)
{
  extern __shared__ __align__(16) char smem[];
  __bf16* lQ = (__bf16*)smem;                                   // 16 x LQ_STRIDE
  float*  lS = (float*)(smem + 16 * LQ_STRIDE * 2);             // 16 x LS_STRIDE
  __bf16* lP = (__bf16*)(smem + 16 * LQ_STRIDE * 2 + 16 * LS_STRIDE * 4);
  float*  red = (float*)(smem + 16 * LQ_STRIDE * 2 + 16 * LS_STRIDE * 4 +
                         16 * LP_STRIDE * 2);                   // 16 x 16

  const int tid  = threadIdx.x;
  const int lane = tid & 31;
  const int w    = tid >> 5;
  const int lr   = lane & 15;
  const int lh   = (lane >> 4) << 3;
  const int b    = blockIdx.y;
  const int q0   = blockIdx.x * 16;
  const int S = 2048, D = 1024;

  // ---- stage 16x1024 Q tile ----
  for (int u = tid; u < 4096; u += 256) {       // 4-wide units
    const int row = u >> 8;
    const int col = (u & 255) << 2;
    *(v4bf*)&lQ[row * LQ_STRIDE + col] =
        *(const v4bf*)&Q[((size_t)(b * S + q0 + row)) * D + col];
  }
  __syncthreads();

  // ---- phase 1: scores; wave w -> key tiles [w*16, w*16+16), 8 accs/group --
#pragma unroll 1
  for (int g = 0; g < 2; ++g) {
    const int ktbase = w * 16 + g * 8;          // first 16-key tile of group
    v8f acc[8];
#pragma unroll
    for (int t = 0; t < 8; ++t) acc[t] = zero8();
    const __bf16* kbase0 = Km + ((size_t)(b * S + ktbase * 16 + lr)) * D + lh;
#pragma unroll 1
    for (int d0 = 0; d0 < 1024; d0 += 32) {
      Frag a;
      const __bf16* pq = &lQ[lr * LQ_STRIDE + d0 + lh];
      a.h[0] = *(const v8bf*)pq;
      a.h[1] = *(const v8bf*)(pq + 16);
#pragma unroll
      for (int t = 0; t < 8; ++t) {
        const __bf16* kb = kbase0 + (size_t)t * 16 * D + d0;
        Frag bf;
        bf.h[0] = *(const v8bf*)kb;
        bf.h[1] = *(const v8bf*)(kb + 16);
        acc[t] = __builtin_amdgcn_wmma_f32_16x16x32_bf16(
            false, a.v, false, bf.v, (short)0, acc[t], false, false);
      }
    }
#pragma unroll
    for (int t = 0; t < 8; ++t) {
      const int colb = (ktbase + t) * 16 + lr;
#pragma unroll
      for (int r = 0; r < 8; ++r)
        lS[(r + lh) * LS_STRIDE + colb] = acc[t][r] * 0.125f; // 1/sqrt(d_k=64)
    }
  }
  __syncthreads();

  // ---- phase 2: exact softmax, each row handled by 16 threads ----
  const int srow = tid >> 4;
  const int seg  = tid & 15;
  float* rowp = &lS[srow * LS_STRIDE];
  float m = -3.0e38f;
  for (int c = seg * 128; c < seg * 128 + 128; ++c) m = fmaxf(m, rowp[c]);
  red[srow * 16 + seg] = m;
  __syncthreads();
  float rm = -3.0e38f;
#pragma unroll
  for (int t = 0; t < 16; ++t) rm = fmaxf(rm, red[srow * 16 + t]);
  __syncthreads();
  float s = 0.0f;
  for (int c = seg * 128; c < seg * 128 + 128; ++c) {
    const float e = __expf(rowp[c] - rm);
    rowp[c] = e;
    s += e;
  }
  red[srow * 16 + seg] = s;
  __syncthreads();
  float rs = 0.0f;
#pragma unroll
  for (int t = 0; t < 16; ++t) rs += red[srow * 16 + t];
  const float inv = 1.0f / rs;
  for (int c = seg * 128; c < seg * 128 + 128; ++c)
    lP[srow * LP_STRIDE + c] = (__bf16)(rowp[c] * inv);
  __syncthreads();

  // ---- phase 3: Ctx = P @ V; wave w owns cols [w*128, w*128+128), 8 accs ---
  {
    v8f acc[8];
#pragma unroll
    for (int nt = 0; nt < 8; ++nt) acc[nt] = zero8();
    const __bf16* vbase0 = Vt + ((size_t)(b * D + w * 128 + lr)) * S + lh;
#pragma unroll 1
    for (int k0 = 0; k0 < 2048; k0 += 32) {
      Frag a;
      const __bf16* pp = &lP[lr * LP_STRIDE + k0 + lh];
      a.h[0] = *(const v8bf*)pp;
      a.h[1] = *(const v8bf*)(pp + 16);
#pragma unroll
      for (int nt = 0; nt < 8; ++nt) {
        const __bf16* vb = vbase0 + (size_t)nt * 16 * S + k0;
        Frag bf;
        bf.h[0] = *(const v8bf*)vb;
        bf.h[1] = *(const v8bf*)(vb + 16);
        acc[nt] = __builtin_amdgcn_wmma_f32_16x16x32_bf16(
            false, a.v, false, bf.v, (short)0, acc[nt], false, false);
      }
    }
#pragma unroll
    for (int nt = 0; nt < 8; ++nt) {
      const int n0 = w * 128 + nt * 16;
#pragma unroll
      for (int r = 0; r < 8; ++r)
        Ctx[((size_t)(b * S + q0 + r + lh)) * D + n0 + lr] = (__bf16)acc[nt][r];
    }
  }
}

// ---------------------------------------------------------------------------
extern "C" void kernel_launch(void* const* d_in, const int* in_sizes, int n_in,
                              void* d_out, int out_size, void* d_ws, size_t ws_size,
                              hipStream_t stream)
{
  const float* X  = (const float*)d_in[0];
  const float* Wq = (const float*)d_in[1];
  const float* Wk = (const float*)d_in[2];
  const float* Wv = (const float*)d_in[3];
  const float* Wo = (const float*)d_in[4];
  float* Out = (float*)d_out;

  const int B = 4, S = 2048, D = 1024, M = B * S;
  const size_t TENSOR_BYTES = (size_t)M * D * 2;   // 16 MB each, bf16

  char* ws = (char*)d_ws;
  __bf16* Qb  = (__bf16*)(ws);
  __bf16* Kb  = (__bf16*)(ws + TENSOR_BYTES);
  __bf16* Vtb = (__bf16*)(ws + 2 * TENSOR_BYTES);
  __bf16* Ctx = (__bf16*)(ws + 3 * TENSOR_BYTES);

  dim3 blk(256);
  dim3 g1(D / 128, M / 128);     // 8 x 64 blocks

  // Q = X Wq, K = X Wk (bf16); Vt = (X Wv)^T per batch (bf16)
  wmma_gemm<false, true,  false><<<g1, blk, 0, stream>>>(X, Wq, Qb,  M, D, D, S);
  wmma_gemm<false, true,  false><<<g1, blk, 0, stream>>>(X, Wk, Kb,  M, D, D, S);
  wmma_gemm<false, true,  true ><<<g1, blk, 0, stream>>>(X, Wv, Vtb, M, D, D, S);

  // fused scores + softmax + P@V
  dim3 g2(S / 16, B);            // 128 x 4 blocks
  attn_fused<<<g2, blk, ATT_SMEM_BYTES, stream>>>(Qb, Kb, Vtb, Ctx);

  // Out = Ctx Wo (fp32 out)
  wmma_gemm<true, false, false><<<g1, blk, 0, stream>>>(Ctx, Wo, Out, M, D, D, S);
}
